// GruRNN_23811298689964
// MI455X (gfx1250) — compile-verified, matrix operand order checked by reading, not run on hardware
//
#include <hip/hip_runtime.h>

// ---------------------------------------------------------------------------
// GRU (2 layers, H=512, B=128, T=1024) + FC, persistent-kernel bf16 WMMA.
// Round 2: split 6-acc GEMM into two 3-acc K-passes + bounded unroll to kill
// the scratch spills seen in round 1 (weights must stream from L2, not spill).
// ---------------------------------------------------------------------------

typedef __attribute__((ext_vector_type(16))) __bf16 v16bf;
typedef __attribute__((ext_vector_type(8)))  float  v8f;
typedef __bf16 bf16_t;

#define HID    512
#define BATCH  128
#define TLEN   1024
#define NBLK   32          // persistent grid: 32 blocks x 256 thr = 256 waves
#define NTHR   256

// ---------------- helpers ----------------

__device__ __forceinline__ v16bf ldv(const bf16_t* p) {
  return *(const v16bf*)p;   // 32B vector load (ds_load / global_load b128 pair)
}

__device__ __forceinline__ v8f wmma_bf16(v16bf a, v16bf b, v8f c) {
  // D = A(16x32 bf16) * B(32x16 bf16) + C(16x16 f32)
  return __builtin_amdgcn_wmma_f32_16x16x32_bf16(
      /*neg_a=*/false, a, /*neg_b=*/false, b,
      /*c_mod=*/(short)0, c, /*reuse_a=*/false, /*reuse_b=*/false);
}

__device__ __forceinline__ float sigf(float x) {
  return 1.0f / (1.0f + __expf(-x));
}
__device__ __forceinline__ float tanhfast(float x) {
  float e = __expf(-2.0f * x);
  return (1.0f - e) / (1.0f + e);
}

// Device-wide barrier: monotonic counter in workspace (zeroed each launch).
__device__ __forceinline__ void grid_barrier(unsigned* bar, unsigned* epoch) {
  __syncthreads();
  if (threadIdx.x == 0) {
    *epoch += gridDim.x;
    __threadfence();
    __hip_atomic_fetch_add(bar, 1u, __ATOMIC_ACQ_REL, __HIP_MEMORY_SCOPE_AGENT);
    while (__hip_atomic_load(bar, __ATOMIC_ACQUIRE, __HIP_MEMORY_SCOPE_AGENT) < *epoch) {
      __builtin_amdgcn_s_sleep(2);
    }
    __threadfence();
  }
  __syncthreads();
}

// Stage 16 rows x 512 cols of bf16 h-state (contiguous) into LDS.
__device__ __forceinline__ void stage_h(bf16_t* lds, const bf16_t* __restrict__ hsrc) {
  for (int i = threadIdx.x * 8; i < 16 * HID; i += NTHR * 8) {
    *(uint4*)(lds + i) = *(const uint4*)(hsrc + i);
  }
}

// Stage 16 rows of x_t (fp32 -> bf16 convert) into LDS; prefetch next step.
__device__ __forceinline__ void stage_x(bf16_t* lds, const float* __restrict__ xg,
                                        int row0, int t) {
  for (int i = threadIdx.x * 4; i < 16 * HID; i += NTHR * 4) {
    int r = i >> 9;
    int c = i & (HID - 1);
    const float* src = xg + ((size_t)(row0 + r) * TLEN + t) * HID + c;
    float4 v = *(const float4*)src;
    lds[i + 0] = (bf16_t)v.x;
    lds[i + 1] = (bf16_t)v.y;
    lds[i + 2] = (bf16_t)v.z;
    lds[i + 3] = (bf16_t)v.w;
    __builtin_prefetch(src + HID, 0, 0);   // next timestep's x (global_prefetch)
  }
}

// 3-gate GEMM pass: acc{r,z,n} += A(16x512 in LDS) x W[chid + g*512][:]^T.
// Bounded unroll keeps B-load addresses loop-variant (no LICM->scratch) and
// live ranges small (3 accs + 1 A + 3 B in flight ~= 64 VGPRs).
__device__ __forceinline__ void gemm3(const bf16_t* __restrict__ ldsA,
                                      const bf16_t* __restrict__ w,
                                      int chid, int lane,
                                      v8f* ar, v8f* az, v8f* an) {
  const int m  = lane & 15;
  const int kh = lane >> 4;
  const bf16_t* __restrict__ A  = ldsA + m * HID + kh * 16;
  const bf16_t* __restrict__ wr = w + (size_t)(chid          ) * HID + kh * 16;
  const bf16_t* __restrict__ wz = w + (size_t)(chid +     HID) * HID + kh * 16;
  const bf16_t* __restrict__ wn = w + (size_t)(chid + 2 * HID) * HID + kh * 16;
  v8f r = *ar, z = *az, n = *an;
#pragma unroll 2
  for (int k0 = 0; k0 < HID; k0 += 32) {
    v16bf a  = ldv(A + k0);
    v16bf b0 = ldv(wr + k0);
    v16bf b1 = ldv(wz + k0);
    v16bf b2 = ldv(wn + k0);
    r = wmma_bf16(a, b0, r);
    z = wmma_bf16(a, b1, z);
    n = wmma_bf16(a, b2, n);
  }
  *ar = r; *az = z; *an = n;
}

// One GRU cell tile: two 3-acc K-passes, fused gate math, write new h tile.
__device__ __forceinline__ void gru_tile(
    const bf16_t* __restrict__ lds_in, const bf16_t* __restrict__ lds_h,
    const bf16_t* __restrict__ w_ih, const bf16_t* __restrict__ w_hh,
    float bir, float biz, float bin, float bhr, float bhz, float bhn,
    bf16_t* __restrict__ hout, int row0, int chid, int lane) {
  const int kh = lane >> 4;
  v8f ir = {}, iz = {}, inn = {};
  v8f hr = {}, hz = {}, hn = {};

  gemm3(lds_in, w_ih, chid, lane, &ir, &iz, &inn);   // input projection
  gemm3(lds_h,  w_hh, chid, lane, &hr, &hz, &hn);    // recurrent projection

#pragma unroll
  for (int e = 0; e < 8; ++e) {
    const int M = e + 8 * kh;                 // C/D layout: vgpr e, lane half
    float r  = sigf(ir[e] + bir + hr[e] + bhr);
    float z  = sigf(iz[e] + biz + hz[e] + bhz);
    float nn = tanhfast(inn[e] + bin + r * (hn[e] + bhn));
    float hp = (float)lds_h[M * HID + chid];
    float hv = (1.0f - z) * nn + z * hp;
    hout[(size_t)(row0 + M) * HID + chid] = (bf16_t)hv;
  }
}

// ---------------- kernels ----------------

__global__ void gru_zero_ws(unsigned* __restrict__ p, int n) {
  int i = blockIdx.x * blockDim.x + threadIdx.x;
  if (i < n) p[i] = 0u;
}

__global__ void gru_cvt_bf16(const float* __restrict__ s, bf16_t* __restrict__ d, int n) {
  int i = blockIdx.x * blockDim.x + threadIdx.x;
  if (i < n) d[i] = (bf16_t)s[i];
}

__global__ void __launch_bounds__(NTHR)
gru_persistent(const float* __restrict__ x,
               const float* __restrict__ bih0, const float* __restrict__ bhh0,
               const float* __restrict__ bih1, const float* __restrict__ bhh1,
               const float* __restrict__ fcb,
               const bf16_t* __restrict__ wih0, const bf16_t* __restrict__ whh0,
               const bf16_t* __restrict__ wih1, const bf16_t* __restrict__ whh1,
               const bf16_t* __restrict__ fcw,
               bf16_t* __restrict__ h0buf,      // 2 x [128,512] bf16
               bf16_t* __restrict__ h1buf,      // 2 x [128,512] bf16
               float* __restrict__ out,         // [128,512] f32
               unsigned* __restrict__ bar) {
  __shared__ __align__(32) bf16_t lds_in[16 * HID];  // 16 KB
  __shared__ __align__(32) bf16_t lds_h [16 * HID];  // 16 KB

  const int lane = threadIdx.x & 31;
  const int wv   = threadIdx.x >> 5;
  const int wid  = blockIdx.x * 8 + wv;     // 0..255
  const int rb   = wid >> 5;                // 0..7 : same for all waves in block
  const int cb   = wid & 31;                // 0..31
  const int row0 = rb * 16;
  const int n    = lane & 15;
  const int kh   = lane >> 4;
  const int chid = cb * 16 + n;             // hidden column owned by this lane

  // Hoist per-lane bias scalars (constant across timesteps).
  const float bir0 = bih0[chid], biz0 = bih0[HID + chid], bin0 = bih0[2 * HID + chid];
  const float bhr0 = bhh0[chid], bhz0 = bhh0[HID + chid], bhn0 = bhh0[2 * HID + chid];
  const float bir1 = bih1[chid], biz1 = bih1[HID + chid], bin1 = bih1[2 * HID + chid];
  const float bhr1 = bhh1[chid], bhz1 = bhh1[HID + chid], bhn1 = bhh1[2 * HID + chid];
  const float fcbl = fcb[chid];

  const size_t HB = (size_t)BATCH * HID;    // elements per h buffer
  unsigned epoch = 0;

  for (int t = 0; t < TLEN; ++t) {
    const int pin  = t & 1;
    const int pout = pin ^ 1;

    // ---- layer 0: h0_new = GRU(x_t, h0_prev) ----
    stage_x(lds_in, x, row0, t);
    stage_h(lds_h, h0buf + (size_t)pin * HB + (size_t)row0 * HID);
    __syncthreads();
    gru_tile(lds_in, lds_h, wih0, whh0,
             bir0, biz0, bin0, bhr0, bhz0, bhn0,
             h0buf + (size_t)pout * HB, row0, chid, lane);
    grid_barrier(bar, &epoch);

    // ---- layer 1: h1_new = GRU(h0_new, h1_prev) ----
    stage_h(lds_in, h0buf + (size_t)pout * HB + (size_t)row0 * HID);
    stage_h(lds_h,  h1buf + (size_t)pin  * HB + (size_t)row0 * HID);
    __syncthreads();
    gru_tile(lds_in, lds_h, wih1, whh1,
             bir1, biz1, bin1, bhr1, bhz1, bhn1,
             h1buf + (size_t)pout * HB, row0, chid, lane);
    grid_barrier(bar, &epoch);
  }

  // ---- FC: out = h1_final @ fc_w^T + fc_b (final parity = 0 after 1024 steps)
  stage_h(lds_h, h1buf + (size_t)row0 * HID);
  __syncthreads();
  {
    const int m = lane & 15;
    const bf16_t* __restrict__ A = lds_h + m * HID + kh * 16;
    const bf16_t* __restrict__ W = fcw + (size_t)chid * HID + kh * 16;
    v8f acc = {};
#pragma unroll 2
    for (int k0 = 0; k0 < HID; k0 += 32) {
      acc = wmma_bf16(ldv(A + k0), ldv(W + k0), acc);
    }
#pragma unroll
    for (int e = 0; e < 8; ++e) {
      const int M = e + 8 * kh;
      out[(size_t)(row0 + M) * HID + chid] = acc[e] + fcbl;
    }
  }
}

// ---------------- host launch ----------------

extern "C" void kernel_launch(void* const* d_in, const int* in_sizes, int n_in,
                              void* d_out, int out_size, void* d_ws, size_t ws_size,
                              hipStream_t stream) {
  (void)in_sizes; (void)n_in; (void)out_size; (void)ws_size;

  const float* x     = (const float*)d_in[0];
  const float* wih0f = (const float*)d_in[1];
  const float* whh0f = (const float*)d_in[2];
  const float* bih0  = (const float*)d_in[3];
  const float* bhh0  = (const float*)d_in[4];
  const float* wih1f = (const float*)d_in[5];
  const float* whh1f = (const float*)d_in[6];
  const float* bih1  = (const float*)d_in[7];
  const float* bhh1  = (const float*)d_in[8];
  const float* fcwf  = (const float*)d_in[9];
  const float* fcb   = (const float*)d_in[10];
  float* out = (float*)d_out;

  char* ws = (char*)d_ws;
  const size_t HB = (size_t)BATCH * HID;          // 65536 elements
  const size_t GW = (size_t)3 * HID * HID;        // 786432 elements (3H x H)

  unsigned* bar   = (unsigned*)ws;                        // 256 B
  bf16_t*   h0buf = (bf16_t*)(ws + 256);                  // 2*HB bf16
  bf16_t*   h1buf = h0buf + 2 * HB;                       // 2*HB bf16
  bf16_t*   wih0q = h1buf + 2 * HB;
  bf16_t*   whh0q = wih0q + GW;
  bf16_t*   wih1q = whh0q + GW;
  bf16_t*   whh1q = wih1q + GW;
  bf16_t*   fcwq  = whh1q + GW;                           // HID*HID bf16

  // Zero barrier + both h double-buffers (bf16 zero == 0x0000). Deterministic
  // per-launch init so graph replays are identical.
  const int zwords = (int)((256 + 4 * HB * sizeof(bf16_t)) / 4);
  gru_zero_ws<<<(zwords + 255) / 256, 256, 0, stream>>>(bar, zwords);

  // One-time (per launch) fp32 -> bf16 weight conversion into workspace.
  const int gw = (int)GW;
  gru_cvt_bf16<<<(gw + 255) / 256, 256, 0, stream>>>(wih0f, wih0q, gw);
  gru_cvt_bf16<<<(gw + 255) / 256, 256, 0, stream>>>(whh0f, whh0q, gw);
  gru_cvt_bf16<<<(gw + 255) / 256, 256, 0, stream>>>(wih1f, wih1q, gw);
  gru_cvt_bf16<<<(gw + 255) / 256, 256, 0, stream>>>(whh1f, whh1q, gw);
  gru_cvt_bf16<<<(HID * HID + 255) / 256, 256, 0, stream>>>(fcwf, fcwq, HID * HID);

  // Persistent recurrent kernel: 32 blocks x 256 threads (co-resident).
  gru_persistent<<<NBLK, NTHR, 0, stream>>>(
      x, bih0, bhh0, bih1, bhh1, fcb,
      wih0q, whh0q, wih1q, whh1q, fcwq,
      h0buf, h1buf, out, bar);
}
